// HypEqualLinear_27659589386536
// MI455X (gfx1250) — compile-verified, matrix operand order checked by reading, not run on hardware
//
#include <hip/hip_runtime.h>

typedef __attribute__((ext_vector_type(16))) _Float16 v16h;
typedef __attribute__((ext_vector_type(8)))  float    v8f;
typedef __attribute__((ext_vector_type(4)))  _Float16 v4h;

#define D_IN   512
#define D_OUT  512
#define BM     64      // rows per workgroup
#define KC     32      // K per chunk
#define NCHUNK (D_IN / KC)
#define EPS_F  1e-5f
#define PROJ_MAXNORM 0.999f   // (1 - 1e-3) / sqrt(c), c = 1
#define SCALE_F 0.044194173824159216f  // 1/sqrt(512)
#define GAIN_F  1.4142135623730951f

__device__ __forceinline__ float artanh_c(float z) {
    z = fminf(fmaxf(z, -1.f + EPS_F), 1.f - EPS_F);
    return 0.5f * logf((1.f + z) / (1.f - z));
}

// ---------------- prep: W * SCALE -> f16 (row-major [O][K]) ----------------
__global__ void prep_weight_f16(const float* __restrict__ w,
                                _Float16* __restrict__ wf16, int n4) {
    int i = blockIdx.x * blockDim.x + threadIdx.x;
    if (i >= n4) return;
    const float4 v = ((const float4*)w)[i];
    v4h h;
    h.x = (_Float16)(v.x * SCALE_F);
    h.y = (_Float16)(v.y * SCALE_F);
    h.z = (_Float16)(v.z * SCALE_F);
    h.w = (_Float16)(v.w * SCALE_F);
    ((v4h*)wf16)[i] = h;
}

// ---------------- prep: bexp = expmap0(bias), y2 = ||bexp||^2 --------------
__global__ void prep_bias_expmap(const float* __restrict__ bias,
                                 float* __restrict__ bexp,
                                 float* __restrict__ y2out) {
    __shared__ float s1, s2;
    int t = threadIdx.x;
    float b = bias[t];               // LR_MUL = 1
    if (t == 0) { s1 = 0.f; s2 = 0.f; }
    __syncthreads();
    atomicAdd(&s1, b * b);
    __syncthreads();
    float un = fmaxf(sqrtf(s1), EPS_F);
    float be = tanhf(un) * b / un;   // sqrt_c = 1
    bexp[t] = be;
    atomicAdd(&s2, be * be);
    __syncthreads();
    if (t == 0) *y2out = s2;
}

// ---------------- main fused kernel ----------------------------------------
// 512 threads = 16 waves. wave w: m_group = w>>2 (16 rows), n_group = w&3 (128 cols)
// Double-buffered LDS: B slab staged via global_load_async_to_lds_b128 (ASYNCcnt),
// A slab converted fp32->f16 in-flight while accumulating ||x||^2.
__global__ __launch_bounds__(512)
void hyp_equal_linear_kernel(const float* __restrict__ x,
                             const _Float16* __restrict__ wf16,
                             const float* __restrict__ bexp,
                             const float* __restrict__ y2ptr,
                             float* __restrict__ out) {
    __shared__ _Float16 ldsA[2][BM * KC];        // 2 x 4 KB  [row][k]
    __shared__ _Float16 ldsB[2][D_OUT * KC];     // 2 x 32 KB [o][k]
    __shared__ float ldsXsq[BM], ldsMxsq[BM], ldsMxb[BM], ldsLsq[BM];
    __shared__ float rowA[BM], rowB[BM], rowG[BM];

    const int tid     = threadIdx.x;
    const int wave    = tid >> 5;
    const int lane    = tid & 31;
    const int half    = lane >> 4;     // 0: lanes 0-15, 1: lanes 16-31
    const int lm      = lane & 15;
    const int m_group = wave >> 2;     // 0..3
    const int n_group = wave & 3;      // 0..3
    const int m0      = blockIdx.x * BM;
    const int row     = tid >> 3;      // A-slab row loaded by this thread
    const int ks      = (tid & 7) * 4; // A-slab k segment (4 floats)

    if (tid < BM) {
        ldsXsq[tid] = 0.f; ldsMxsq[tid] = 0.f; ldsMxb[tid] = 0.f; ldsLsq[tid] = 0.f;
    }
    __syncthreads();

    // -------- prologue: stage chunk 0 into buffer 0 -----------------------
    {
        const float4 xv = *(const float4*)(x + (size_t)(m0 + row) * D_IN + ks);
        atomicAdd(&ldsXsq[row], xv.x * xv.x + xv.y * xv.y + xv.z * xv.z + xv.w * xv.w);
        v4h h;
        h.x = (_Float16)xv.x; h.y = (_Float16)xv.y;
        h.z = (_Float16)xv.z; h.w = (_Float16)xv.w;
        *(v4h*)(&ldsA[0][row * KC + ks]) = h;

        const char* gB = (const char*)(wf16 + (size_t)tid * D_IN);
        const unsigned la = (unsigned)(size_t)&ldsB[0][tid * KC];
        #pragma unroll
        for (int i = 0; i < 4; ++i)
            asm volatile("global_load_async_to_lds_b128 %0, %1, off"
                         :: "v"(la + 16u * i), "v"(gB + 16 * i) : "memory");
    }

    v8f acc[8] = {};

    // -------- K loop: compute buf(cur) while DMA fills buf(nxt) -----------
    for (int kc = 0; kc < NCHUNK; ++kc) {
        const int cur = kc & 1;
        const int nxt = cur ^ 1;

        asm volatile("s_wait_asynccnt 0x0" ::: "memory");
        __syncthreads();

        float4 xv{};
        const bool have_nxt = (kc + 1 < NCHUNK);
        if (have_nxt) {
            const char* gB = (const char*)(wf16 + (size_t)tid * D_IN + (kc + 1) * KC);
            const unsigned la = (unsigned)(size_t)&ldsB[nxt][tid * KC];
            #pragma unroll
            for (int i = 0; i < 4; ++i)
                asm volatile("global_load_async_to_lds_b128 %0, %1, off"
                             :: "v"(la + 16u * i), "v"(gB + 16 * i) : "memory");
            xv = *(const float4*)(x + (size_t)(m0 + row) * D_IN + (kc + 1) * KC + ks);
        }

        // A fragment (16x32 f16): lane layout per ISA 7.12.2
        const _Float16* ar = &ldsA[cur][(m_group * 16 + lm) * KC];
        v16h afrag;
        ((uint4*)&afrag)[0] = *(const uint4*)(ar + half * 8);
        ((uint4*)&afrag)[1] = *(const uint4*)(ar + 16 + half * 8);

        #pragma unroll
        for (int t = 0; t < 8; ++t) {
            const _Float16* br = &ldsB[cur][(n_group * 128 + t * 16 + lm) * KC];
            v16h bfrag;
            ((uint4*)&bfrag)[0] = *(const uint4*)(br + half * 8);
            ((uint4*)&bfrag)[1] = *(const uint4*)(br + 16 + half * 8);
            acc[t] = __builtin_amdgcn_wmma_f32_16x16x32_f16(
                false, afrag, false, bfrag, (short)0, acc[t], false, false);
        }

        if (have_nxt) {
            atomicAdd(&ldsXsq[row], xv.x * xv.x + xv.y * xv.y + xv.z * xv.z + xv.w * xv.w);
            v4h h;
            h.x = (_Float16)xv.x; h.y = (_Float16)xv.y;
            h.z = (_Float16)xv.z; h.w = (_Float16)xv.w;
            *(v4h*)(&ldsA[nxt][row * KC + ks]) = h;
        }
    }

    // -------- epilogue phase 1: reduce ||mx||^2 and mx.b per row ----------
    float bv[8];
    #pragma unroll
    for (int t = 0; t < 8; ++t) bv[t] = bexp[n_group * 128 + t * 16 + lm];

    #pragma unroll
    for (int r = 0; r < 8; ++r) {
        float psq = 0.f, pxb = 0.f;
        #pragma unroll
        for (int t = 0; t < 8; ++t) {
            const float v = acc[t][r];
            psq += v * v;
            pxb += v * bv[t];
        }
        #pragma unroll
        for (int m = 1; m < 16; m <<= 1) {
            psq += __shfl_xor(psq, m, 32);
            pxb += __shfl_xor(pxb, m, 32);
        }
        if (lm == 0) {
            const int ml = m_group * 16 + r + half * 8;
            atomicAdd(&ldsMxsq[ml], psq);
            atomicAdd(&ldsMxb[ml], pxb);
        }
    }
    __syncthreads();

    // -------- phase 2: per-row scalars (mobius_matvec/add/project) --------
    if (tid < BM) {
        const float xsq  = ldsXsq[tid];
        const float mxsq = ldsMxsq[tid];
        const float mxb  = ldsMxb[tid];
        const float y2   = *y2ptr;
        const float xn   = fmaxf(sqrtf(xsq), EPS_F);
        const float mxn  = fmaxf(sqrtf(mxsq), EPS_F);
        float f = tanhf(mxn / xn * artanh_c(xn)) / mxn;   // sqrt_c = 1
        if (mxsq == 0.f) f = 0.f;                          // zero_row guard
        const float x2m = f * f * mxsq;                    // ||mv||^2
        const float xy  = f * mxb;                         // mv . b
        const float den = 1.f + 2.f * xy + x2m * y2 + EPS_F;
        const float cA  = (1.f + 2.f * xy + y2) / den;     // coef on mv
        const float cB  = (1.f - x2m) / den;               // coef on b
        const float n2  = cA * cA * x2m + 2.f * cA * cB * xy + cB * cB * y2;
        const float nn  = fmaxf(sqrtf(n2), EPS_F);
        const float ps  = (nn > PROJ_MAXNORM) ? (PROJ_MAXNORM / nn) : 1.f;
        rowA[tid] = ps * cA * f;   // applied to raw mx accumulator
        rowB[tid] = ps * cB;       // applied to bexp element
    }
    __syncthreads();

    // -------- phase 3: mobius_add + project + leaky_relu; reduce norm -----
    float av[8], bc[8];
    #pragma unroll
    for (int r = 0; r < 8; ++r) {
        const int ml = m_group * 16 + r + half * 8;
        av[r] = rowA[ml];
        bc[r] = rowB[ml];
    }
    #pragma unroll
    for (int r = 0; r < 8; ++r) {
        float lsq = 0.f;
        #pragma unroll
        for (int t = 0; t < 8; ++t) {
            float v = av[r] * acc[t][r] + bc[r] * bv[t];
            v = (v < 0.f) ? 0.2f * v : v;
            acc[t][r] = v;
            lsq += v * v;
        }
        #pragma unroll
        for (int m = 1; m < 16; m <<= 1) lsq += __shfl_xor(lsq, m, 32);
        if (lm == 0) atomicAdd(&ldsLsq[m_group * 16 + r + half * 8], lsq);
    }
    __syncthreads();

    // -------- phase 4: mobius_scalar_mult gain per row --------------------
    if (tid < BM) {
        const float ln = fmaxf(sqrtf(ldsLsq[tid]), EPS_F);
        rowG[tid] = tanhf(GAIN_F * artanh_c(ln)) / ln;     // sqrt_c = 1
    }
    __syncthreads();

    // -------- phase 5: scale + store --------------------------------------
    float gv[8];
    #pragma unroll
    for (int r = 0; r < 8; ++r) gv[r] = rowG[m_group * 16 + r + half * 8];

    #pragma unroll
    for (int t = 0; t < 8; ++t) {
        const int col = n_group * 128 + t * 16 + lm;
        #pragma unroll
        for (int r = 0; r < 8; ++r) {
            const int ml = m_group * 16 + r + half * 8;
            out[(size_t)(m0 + ml) * D_OUT + col] = gv[r] * acc[t][r];
        }
    }
}

extern "C" void kernel_launch(void* const* d_in, const int* in_sizes, int n_in,
                              void* d_out, int out_size, void* d_ws, size_t ws_size,
                              hipStream_t stream) {
    const float* x      = (const float*)d_in[0];
    const float* weight = (const float*)d_in[1];
    const float* bias   = (const float*)d_in[2];
    float* out = (float*)d_out;

    const int nrows = in_sizes[0] / D_IN;        // 131072
    const int wnum  = in_sizes[1];               // 262144

    // workspace layout: [0, 512KB) f16 weight; [512KB, +2KB) bexp; then y2
    _Float16* wf16 = (_Float16*)d_ws;
    float* bexp    = (float*)((char*)d_ws + (size_t)D_OUT * D_IN * sizeof(_Float16));
    float* y2ptr   = bexp + D_OUT;

    prep_weight_f16<<<(wnum / 4 + 255) / 256, 256, 0, stream>>>(weight, wf16, wnum / 4);
    prep_bias_expmap<<<1, D_OUT, 0, stream>>>(bias, bexp, y2ptr);
    hyp_equal_linear_kernel<<<nrows / BM, 512, 0, stream>>>(x, wf16, bexp, y2ptr, out);
}
